// PAM_Module_90237262889330
// MI455X (gfx1250) — matrix operand order repaired and checked by reading, not compiled, not run on hardware
//
#include <hip/hip_runtime.h>

// PAM module on MI455X: WMMA f16 projections + fused flash-attention.
// Whole working set (~100MB) fits in 192MB L2 -> compute bound -> everything
// heavy goes through v_wmma_f32_16x16x32_f16.  LDS staging uses CDNA5
// global_load_async_to_lds (ASYNCcnt) with double-buffering so the async
// copies overlap the WMMA stage instead of serializing on s_wait_asynccnt 0.

#define C_IN  512
#define C_QK  64
#define NPIX  9216   // 96*96
#define BATCH 2
#define NKEY  32     // key block per attention iteration

typedef __attribute__((ext_vector_type(16))) _Float16 v16h;
typedef __attribute__((ext_vector_type(8)))  float    v8f;

union AFrag { v16h h; unsigned int u[8]; };

static __device__ inline v8f wmma_f16(const AFrag& a, const AFrag& b, v8f c) {
    return __builtin_amdgcn_wmma_f32_16x16x32_f16(
        /*neg_a=*/false, a.h, /*neg_b=*/false, b.h,
        /*c_mod=*/(short)0, c, /*reuse_a=*/false, /*reuse_b=*/false);
}

// ---- CDNA5 async global->LDS copy (16B per lane), guarded ------------------
typedef int v4i __attribute__((vector_size(16)));
typedef __attribute__((address_space(1))) v4i* gptr_v4i;
typedef __attribute__((address_space(3))) v4i* lptr_v4i;

static __device__ inline void async_copy16(const void* g, void* l) {
#if __has_builtin(__builtin_amdgcn_global_load_async_to_lds_b128)
    __builtin_amdgcn_global_load_async_to_lds_b128(
        (gptr_v4i)g, (lptr_v4i)l, 0, 0);
#else
    *(uint4*)l = *(const uint4*)g;
#endif
}

// Wait until at most N async ops outstanding (async loads complete in order,
// so N = ops-in-flight-for-next-chunk guarantees the current chunk landed).
template <int N>
static __device__ inline void wait_async_le() {
#if __has_builtin(__builtin_amdgcn_s_wait_asynccnt)
    __builtin_amdgcn_s_wait_asynccnt(N);
#endif
}

// ---------------------------------------------------------------------------
// Kernel 0a: convert conv weights fp32 -> f16 (row-major unchanged).
// ---------------------------------------------------------------------------
__global__ __launch_bounds__(256) void cvt_w(
    const float* __restrict__ wq, const float* __restrict__ wk,
    const float* __restrict__ wv,
    _Float16* __restrict__ wq16, _Float16* __restrict__ wk16,
    _Float16* __restrict__ wv16)
{
    const int i = blockIdx.x * 256 + threadIdx.x;
    if (i < C_QK * C_IN) {
        wq16[i] = (_Float16)wq[i];
        wk16[i] = (_Float16)wk[i];
    }
    for (int j = i; j < C_IN * C_IN; j += (int)gridDim.x * 256)
        wv16[j] = (_Float16)wv[j];
}

// ---------------------------------------------------------------------------
// Kernel 0b: tiled transpose+convert  x[b][c][n] f32 -> x16t[b][n][c] f16.
// ---------------------------------------------------------------------------
__global__ __launch_bounds__(256) void cvt_x(
    const float* __restrict__ x, _Float16* __restrict__ x16t)
{
    const int b  = blockIdx.z;
    const int n0 = blockIdx.x * 64;
    const int c0 = blockIdx.y * 64;
    __shared__ __align__(16) _Float16 tile[64 * 64];   // [n][c]
    const int t = threadIdx.x;
    #pragma unroll
    for (int i = 0; i < 16; ++i) {
        int li = t + 256 * i;            // 0..4095
        int cc = li >> 6, nn = li & 63;
        tile[nn * 64 + cc] =
            (_Float16)x[((size_t)(b * C_IN + c0 + cc)) * NPIX + n0 + nn];
    }
    __syncthreads();
    #pragma unroll
    for (int i = 0; i < 2; ++i) {
        int li = t + 256 * i;            // 0..511
        int nn = li >> 3, part = li & 7;
        *(uint4*)(x16t + ((size_t)(b * NPIX + n0 + nn)) * C_IN + c0 + part * 8) =
            *(const uint4*)(tile + nn * 64 + part * 8);
    }
}

// ---------------------------------------------------------------------------
// Kernel 1: WMMA projection GEMM.  Y[640 x N] = W[640 x 512] * X[512 x N].
// grid.y: 0..7 -> V row-blocks, 8 -> Q, 9 -> K.  Block tile 64x64, K=512.
// Double-buffered async LDS staging of 64x32 W/X chunks.
// ---------------------------------------------------------------------------
__global__ __launch_bounds__(256) void proj_gemm(
    const _Float16* __restrict__ x16t,           // [b][n][512]
    const _Float16* __restrict__ wq16, const float* __restrict__ bq,
    const _Float16* __restrict__ wk16, const float* __restrict__ bk,
    const _Float16* __restrict__ wv16, const float* __restrict__ bv,
    _Float16* __restrict__ q16, _Float16* __restrict__ k16,
    _Float16* __restrict__ v16)
{
    const int b    = blockIdx.z;
    const int n0   = blockIdx.x * 64;
    const int oblk = blockIdx.y;          // 0..7: V, 8: Q, 9: K

    const _Float16* W;
    const float* bias;
    int mode, o0;
    if (oblk < 8)       { W = wv16 + (size_t)oblk * 64 * C_IN; bias = bv + oblk * 64; mode = 0; o0 = oblk * 64; }
    else if (oblk == 8) { W = wq16; bias = bq; mode = 1; o0 = 0; }
    else                { W = wk16; bias = bk; mode = 2; o0 = 0; }

    __shared__ __align__(16) _Float16 wLds[2][64 * 32];   // [o][c]
    __shared__ __align__(16) _Float16 xLds[2][64 * 32];   // [n][c]

    const int t    = threadIdx.x;
    const int lane = t & 31;
    const int wave = t >> 5;
    const int lm   = lane & 15;
    const int lh   = lane >> 4;
    const int otile  = wave & 3;          // 0..3
    const int ntile0 = (wave >> 2) * 2;   // 0 or 2

    const int row = t >> 2, part = t & 3;     // 64 rows x 4 x 16B
    const _Float16* wsrc = W + (size_t)row * C_IN + part * 8;
    const _Float16* xsrc =
        x16t + ((size_t)(b * NPIX + n0 + row)) * C_IN + part * 8;
    _Float16* wdst = (_Float16*)wLds[0] + row * 32 + part * 8; // buf stride below
    _Float16* xdst = (_Float16*)xLds[0] + row * 32 + part * 8;
    const int bufStride = 64 * 32;

    v8f acc0 = {}, acc1 = {};
    const int NCH = C_IN / 32;            // 16 K-chunks

    // prologue: stage chunk 0 into buffer 0
    async_copy16(wsrc, wdst);
    async_copy16(xsrc, xdst);

    for (int c = 0; c < NCH; ++c) {
        const int cur = c & 1;
        if (c + 1 < NCH) {
            const int nxt = cur ^ 1;
            async_copy16(wsrc + (c + 1) * 32, wdst + nxt * bufStride);
            async_copy16(xsrc + (c + 1) * 32, xdst + nxt * bufStride);
            wait_async_le<2>();           // chunk c landed; chunk c+1 in flight
        } else {
            wait_async_le<0>();
        }
        __syncthreads();

        AFrag a;
        {
            const unsigned int* wr =
                (const unsigned int*)(wLds[cur] + (otile * 16 + lm) * 32);
            #pragma unroll
            for (int v = 0; v < 4; ++v) {
                a.u[v]     = wr[8 * lh + v];        // K = 16*lh + 2v
                a.u[4 + v] = wr[8 * lh + 4 + v];    // K = 16*lh + 8 + 2v
            }
        }
        AFrag b0, b1;
        {
            const unsigned int* xr0 =
                (const unsigned int*)(xLds[cur] + (ntile0 * 16 + lm) * 32);
            const unsigned int* xr1 =
                (const unsigned int*)(xLds[cur] + ((ntile0 + 1) * 16 + lm) * 32);
            #pragma unroll
            for (int v = 0; v < 8; ++v) {
                b0.u[v] = xr0[8 * lh + v];          // K = 16*lh + 2v
                b1.u[v] = xr1[8 * lh + v];
            }
        }
        acc0 = wmma_f16(a, b0, acc0);
        acc1 = wmma_f16(a, b1, acc1);
        __syncthreads();   // all reads of buf cur done before it is re-staged
    }

    // epilogue: add bias, store in consumer layouts
    #pragma unroll
    for (int ti = 0; ti < 2; ++ti) {
        const v8f& acc = ti ? acc1 : acc0;
        const int nl = (ntile0 + ti) * 16 + lm;     // n within 64-block
        #pragma unroll
        for (int r = 0; r < 8; ++r) {
            const int ol = otile * 16 + r + 8 * lh; // o within 64-block
            const float val = acc[r] + bias[ol];
            if (mode == 0)
                v16[((size_t)(b * C_IN + o0 + ol)) * NPIX + n0 + nl] = (_Float16)val;
            else if (mode == 1)
                q16[((size_t)(b * NPIX + n0 + nl)) * C_QK + ol] = (_Float16)val;
            else
                k16[((size_t)(b * NPIX + n0 + nl)) * C_QK + ol] = (_Float16)val;
        }
    }
}

// ---------------------------------------------------------------------------
// Kernel 2: flash attention + PV + residual.
// Workgroup = 8 waves = 256 threads per 16-query block.
// Double-buffered async staging of K/V tiles (9 async ops per stage).
// ---------------------------------------------------------------------------
__global__ __launch_bounds__(256) void attn_kernel(
    const _Float16* __restrict__ q16, const _Float16* __restrict__ k16,
    const _Float16* __restrict__ v16, const float* __restrict__ xg,
    const float* __restrict__ gamma_p, float* __restrict__ out)
{
    const int b  = blockIdx.y;
    const int q0 = blockIdx.x * 16;

    __shared__ __align__(16) _Float16 kTile[2][NKEY * C_QK];   // [key][d]   2x4 KB
    __shared__ __align__(16) _Float16 vTile[2][C_IN * NKEY];   // [ch][key] 2x32 KB
    __shared__ float    sTile[16 * NKEY];                      // energy      2 KB
    __shared__ __align__(8) _Float16 pTile[16 * NKEY];         // probs       1 KB
    __shared__ float    mRow[16], lRow[16], rRow[16];

    const int t    = threadIdx.x;
    const int lane = t & 31;
    const int wave = t >> 5;
    const int lm   = lane & 15;
    const int lh   = lane >> 4;

    // Q fragments (A layout, d 0..31 and 32..63), resident in VGPRs
    AFrag qa0, qa1;
    {
        const unsigned int* q32 =
            (const unsigned int*)(q16 + ((size_t)(b * NPIX + q0 + lm)) * C_QK);
        #pragma unroll
        for (int v = 0; v < 4; ++v) {
            qa0.u[v]     = q32[8 * lh + v];
            qa0.u[4 + v] = q32[8 * lh + 4 + v];
            qa1.u[v]     = q32[16 + 8 * lh + v];
            qa1.u[4 + v] = q32[20 + 8 * lh + v];
        }
    }

    // stage key-block starting at j0 into LDS buffer `buf` (9 async ops/thread)
    auto stage = [&](int j0, int buf) {
        async_copy16(k16 + ((size_t)(b * NPIX + j0)) * C_QK + t * 8,
                     kTile[buf] + t * 8);
        #pragma unroll
        for (int i = 0; i < 8; ++i) {
            int idx  = t + 256 * i;           // 0..2047
            int ch   = idx >> 2;
            int quad = idx & 3;
            async_copy16(v16 + ((size_t)(b * C_IN + ch)) * NPIX + j0 + quad * 8,
                         vTile[buf] + ch * NKEY + quad * 8);
        }
    };

    v8f acc[4] = {};
    if (t < 16) { mRow[t] = -1e30f; lRow[t] = 0.0f; }

    stage(0, 0);                          // prologue
    const int NBLK = NPIX / NKEY;         // 288

    for (int blk = 0; blk < NBLK; ++blk) {
        const int cur = blk & 1;
        if (blk + 1 < NBLK) {
            stage((blk + 1) * NKEY, cur ^ 1);
            wait_async_le<9>();           // block blk landed; blk+1 in flight
        } else {
            wait_async_le<0>();
        }
        __syncthreads();

        // ---- energy S = Q x K^T (waves 0,1) --------------------------------
        if (wave < 2) {
            const unsigned int* krow =
                (const unsigned int*)(kTile[cur] + (16 * wave + lm) * C_QK);
            AFrag kb0, kb1;
            #pragma unroll
            for (int v = 0; v < 8; ++v) {
                kb0.u[v] = krow[8 * lh + v];
                kb1.u[v] = krow[16 + 8 * lh + v];
            }
            v8f s = {};
            s = wmma_f16(qa0, kb0, s);
            s = wmma_f16(qa1, kb1, s);
            #pragma unroll
            for (int r = 0; r < 8; ++r)
                sTile[(r + 8 * lh) * NKEY + 16 * wave + lm] = s[r];
        }
        __syncthreads();

        // ---- online softmax (wave 0, lanes 0..15) --------------------------
        if (t < 16) {
            const int m = t;
            float mo = mRow[m], mx = mo;
            #pragma unroll 4
            for (int j = 0; j < NKEY; ++j) mx = fmaxf(mx, sTile[m * NKEY + j]);
            float rf = __expf(mo - mx);
            float sum = 0.0f;
            #pragma unroll 4
            for (int j = 0; j < NKEY; ++j) {
                float p = __expf(sTile[m * NKEY + j] - mx);
                sum += p;
                pTile[m * NKEY + j] = (_Float16)p;
            }
            mRow[m] = mx;
            lRow[m] = lRow[m] * rf + sum;
            rRow[m] = rf;
        }
        __syncthreads();

        // ---- rescale accumulators + PV (all 8 waves) -----------------------
        AFrag pa;
        {
            const unsigned int* prow = (const unsigned int*)(pTile + lm * NKEY);
            #pragma unroll
            for (int v = 0; v < 4; ++v) {
                pa.u[v]     = prow[8 * lh + v];
                pa.u[4 + v] = prow[8 * lh + 4 + v];
            }
        }
        #pragma unroll
        for (int r = 0; r < 8; ++r) {
            float rf = rRow[r + 8 * lh];
            #pragma unroll
            for (int i = 0; i < 4; ++i) acc[i][r] *= rf;
        }
        #pragma unroll
        for (int i = 0; i < 4; ++i) {
            const int ch = (wave * 4 + i) * 16 + lm;
            const unsigned int* vrow =
                (const unsigned int*)(vTile[cur] + ch * NKEY);
            AFrag vb;
            #pragma unroll
            for (int v = 0; v < 8; ++v) vb.u[v] = vrow[8 * lh + v];
            acc[i] = wmma_f16(pa, vb, acc[i]);
        }
        __syncthreads();   // all reads of buf cur done before it is re-staged
    }

    // ---- epilogue: out = gamma * (acc / l) + x -----------------------------
    const float g = gamma_p[0];
    #pragma unroll
    for (int i = 0; i < 4; ++i) {
        const int ch = (wave * 4 + i) * 16 + lm;
        #pragma unroll
        for (int r = 0; r < 8; ++r) {
            const int m  = r + 8 * lh;
            const float linv = 1.0f / lRow[m];
            const size_t off = ((size_t)(b * C_IN + ch)) * NPIX + q0 + m;
            out[off] = g * (acc[i][r] * linv) + xg[off];
        }
    }
}

// ---------------------------------------------------------------------------
extern "C" void kernel_launch(void* const* d_in, const int* in_sizes, int n_in,
                              void* d_out, int out_size, void* d_ws, size_t ws_size,
                              hipStream_t stream) {
    const float* x     = (const float*)d_in[0];
    const float* wq    = (const float*)d_in[1];
    const float* bq    = (const float*)d_in[2];
    const float* wk    = (const float*)d_in[3];
    const float* bk    = (const float*)d_in[4];
    const float* wv    = (const float*)d_in[5];
    const float* bv    = (const float*)d_in[6];
    const float* gamma = (const float*)d_in[7];
    float* out = (float*)d_out;

    // workspace (f16): q16/k16 [B][N][64], v16 [B][512][N], x16t [B][N][512],
    //                  wq16/wk16 [64][512], wv16 [512][512]   (~43 MB total)
    _Float16* q16  = (_Float16*)d_ws;
    _Float16* k16  = q16  + (size_t)BATCH * NPIX * C_QK;
    _Float16* v16  = k16  + (size_t)BATCH * NPIX * C_QK;
    _Float16* x16t = v16  + (size_t)BATCH * C_IN * NPIX;
    _Float16* wq16 = x16t + (size_t)BATCH * NPIX * C_IN;
    _Float16* wk16 = wq16 + (size_t)C_QK * C_IN;
    _Float16* wv16 = wk16 + (size_t)C_QK * C_IN;

    cvt_w<<<(C_IN * C_IN) / 256, 256, 0, stream>>>(wq, wk, wv, wq16, wk16, wv16);

    dim3 gx(NPIX / 64, C_IN / 64, BATCH);           // 144 x 8 x 2
    cvt_x<<<gx, 256, 0, stream>>>(x, x16t);

    dim3 gp(NPIX / 64, 10, BATCH);                  // 144 x 10 x 2
    proj_gemm<<<gp, 256, 0, stream>>>(x16t, wq16, bq, wk16, bk, wv16, bv,
                                      q16, k16, v16);

    dim3 ga(NPIX / 16, BATCH);                      // 576 x 2
    attn_kernel<<<ga, 256, 0, stream>>>(q16, k16, v16, x, gamma, out);
}
